// Block_37546604101900
// MI455X (gfx1250) — matrix operand order repaired and checked by reading, not compiled, not run on hardware
//
#include <hip/hip_runtime.h>
#include <math.h>

// ---------------------------------------------------------------------------
// Transformer block for MI455X (gfx1250, wave32, WMMA).
// All heavy GEMMs run on v_wmma_f32_16x16x32_f16 (f16 A/B, f32 accumulate).
// ---------------------------------------------------------------------------

typedef __attribute__((ext_vector_type(16))) _Float16 v16h;
typedef __attribute__((ext_vector_type(8)))  _Float16 v8h;
typedef __attribute__((ext_vector_type(8)))  float    v8f;

#define EMBED 1024
#define SEQ   4096
#define BATCH 4
#define HEADS 16
#define HDIM  64
#define DFF   4096
#define ROWS  (BATCH * SEQ)          // 16384 token rows
#define LN_EPS 1e-5f

// ------------------------------- prep kernels ------------------------------

// Wt[n*K + k] = (f16) W[k*N + n]   (transpose + downconvert, coalesced store)
__global__ void k_transpose_f16(const float* __restrict__ W,
                                _Float16* __restrict__ Wt, int K, int N) {
  long idx = (long)blockIdx.x * blockDim.x + threadIdx.x;
  long total = (long)K * N;
  if (idx >= total) return;
  int n = (int)(idx / K);
  int k = (int)(idx - (long)n * K);
  Wt[idx] = (_Float16)W[(long)k * N + n];
}

__global__ void k_f32_to_f16(const float* __restrict__ a,
                             _Float16* __restrict__ b, long n) {
  long idx = (long)blockIdx.x * blockDim.x + threadIdx.x;
  if (idx < n) b[idx] = (_Float16)a[idx];
}

// ------------------------------- WMMA GEMM ---------------------------------
// C[M,N] = A[M,K] * Bt[N,K]^T (+bias), epilogue selected by `mode`.
// 128x128 block tile, 256 threads = 8 waves (2x4), each wave 4x2 WMMA tiles.

enum { MODE_F32 = 0, MODE_F16 = 1, MODE_PERM = 2, MODE_RELU16 = 3 };

#define LDP 48   // padded LDS pitch in halves (96B, keeps 16B alignment)

__global__ __launch_bounds__(256)
void k_gemm_wmma(const _Float16* __restrict__ A, const _Float16* __restrict__ Bt,
                 int M, int N, int K, const float* __restrict__ bias, int mode,
                 float* __restrict__ outF, _Float16* __restrict__ outH)
{
  __shared__ __align__(32) _Float16 As[128 * LDP];
  __shared__ __align__(32) _Float16 Bs[128 * LDP];

  const int row0 = blockIdx.y * 128;
  const int col0 = blockIdx.x * 128;
  const int t    = threadIdx.x;
  const int lane = t & 31;
  const int wid  = t >> 5;
  const int wm   = wid >> 2;     // wave row   0..1  (64 rows each)
  const int wn   = wid & 3;      // wave col   0..3  (32 cols each)
  const int lg   = lane >> 4;    // lane group 0..1
  const int lr   = lane & 15;

  const v8f vzero = {0.f,0.f,0.f,0.f,0.f,0.f,0.f,0.f};
  v8f acc[4][2];
  #pragma unroll
  for (int i = 0; i < 4; ++i)
    #pragma unroll
    for (int j = 0; j < 2; ++j) acc[i][j] = vzero;

  // staging: thread t copies one 32B chunk (16 halves) per tile per matrix
  const int sr = t >> 1;
  const int sc = (t & 1) << 4;     // 0 or 16 halves

  for (int k0 = 0; k0 < K; k0 += 32) {
    *(v16h*)(&As[sr * LDP + sc]) =
        *(const v16h*)(A  + (size_t)(row0 + sr) * K + k0 + sc);
    *(v16h*)(&Bs[sr * LDP + sc]) =
        *(const v16h*)(Bt + (size_t)(col0 + sr) * K + k0 + sc);
    __syncthreads();

    v16h af[4], bf[2];
    // A fragment (ISA 16-bit A 16x32 layout): lane(m=lr, g=lg) holds
    // halves[0..7] = K 8g..8g+7, halves[8..15] = K 16+8g..23+8g
    #pragma unroll
    for (int mt = 0; mt < 4; ++mt) {
      const _Float16* ap = &As[(wm * 64 + mt * 16 + lr) * LDP];
      v8h lo = *(const v8h*)(ap + 8 * lg);
      v8h hi = *(const v8h*)(ap + 16 + 8 * lg);
      af[mt] = __builtin_shufflevector(lo, hi,
               0,1,2,3,4,5,6,7,8,9,10,11,12,13,14,15);
    }
    // B fragment: lane(n=lr, g=lg) holds halves[0..7] = K 8g..8g+7,
    // halves[8..15] = K 16+8g..23+8g
    #pragma unroll
    for (int nt = 0; nt < 2; ++nt) {
      const _Float16* bp = &Bs[(wn * 32 + nt * 16 + lr) * LDP];
      v8h blo = *(const v8h*)(bp + 8 * lg);
      v8h bhi = *(const v8h*)(bp + 16 + 8 * lg);
      bf[nt] = __builtin_shufflevector(blo, bhi,
               0,1,2,3,4,5,6,7,8,9,10,11,12,13,14,15);
    }

    #pragma unroll
    for (int mt = 0; mt < 4; ++mt)
      #pragma unroll
      for (int nt = 0; nt < 2; ++nt)
        acc[mt][nt] = __builtin_amdgcn_wmma_f32_16x16x32_f16(
            false, af[mt], false, bf[nt], (short)0, acc[mt][nt], false, false);
    __syncthreads();
  }

  // epilogue: C element e of v8f -> m = e + 8*lg, n = lr (per ISA C layout)
  #pragma unroll
  for (int mt = 0; mt < 4; ++mt) {
    #pragma unroll
    for (int nt = 0; nt < 2; ++nt) {
      const int mbase = row0 + wm * 64 + mt * 16 + lg * 8;
      const int col   = col0 + wn * 32 + nt * 16 + lr;
      const float bv  = bias ? bias[col] : 0.0f;
      #pragma unroll
      for (int e = 0; e < 8; ++e) {
        const int row = mbase + e;
        float val = acc[mt][nt][e] + bv;
        if (mode == MODE_F32) {
          outF[(size_t)row * N + col] = val;
        } else if (mode == MODE_F16) {
          outH[(size_t)row * N + col] = (_Float16)val;
        } else if (mode == MODE_RELU16) {
          outH[(size_t)row * N + col] = (_Float16)fmaxf(val, 0.0f);
        } else { // MODE_PERM: [N,S,H*D] -> [N,H,S,D]  (N == EMBED)
          const int b = row >> 12, s = row & (SEQ - 1);
          const int h = col >> 6,  d = col & (HDIM - 1);
          outF[(((size_t)(b * HEADS + h)) * SEQ + s) * HDIM + d] = val;
        }
      }
    }
  }
}

// ------------------------------ attention ----------------------------------
// E[n,qh,kh,s] = sum_d Q[n,s,qh*64+d] * K[n,kh? -> K_out[n,kh,s,d]]
// Vs[n,s,d]    = sum_h V_out[n,h,s,d]
__global__ __launch_bounds__(256)
void k_energy(const _Float16* __restrict__ Q16, const float* __restrict__ Kout,
              const float* __restrict__ Vout, float* __restrict__ E,
              float* __restrict__ Vs)
{
  const int ns = blockIdx.x;                 // n*SEQ + s
  const int n = ns >> 12, s = ns & (SEQ - 1);
  __shared__ float sQ[EMBED];
  __shared__ float sK[EMBED];
  const int t = threadIdx.x;

  const _Float16* qrow = Q16 + (size_t)ns * EMBED;
  for (int i = t; i < EMBED; i += 256) sQ[i] = (float)qrow[i];
  {
    const int h = t >> 4, d4 = (t & 15) << 2;
    const float* kp = Kout + ((size_t)(n * HEADS + h) * SEQ + s) * HDIM + d4;
    #pragma unroll
    for (int j = 0; j < 4; ++j) sK[h * HDIM + d4 + j] = kp[j];
  }
  __syncthreads();

  const int qh = t >> 4, kh = t & 15;
  float e = 0.f;
  #pragma unroll 8
  for (int d = 0; d < HDIM; ++d) e += sQ[qh * HDIM + d] * sK[kh * HDIM + d];
  E[(((size_t)(n * HEADS + qh)) * HEADS + kh) * SEQ + s] = e;

  if (t < HDIM) {
    float v = 0.f;
    for (int h = 0; h < HEADS; ++h)
      v += Vout[((size_t)(n * HEADS + h) * SEQ + s) * HDIM + t];
    Vs[(size_t)ns * HDIM + t] = v;
  }
}

// per-channel (n,qh,kh) softmax stats over s: m = max(E/8), p = sum exp(E/8 - m)
__global__ __launch_bounds__(256)
void k_sm_stats(const float* __restrict__ E, float* __restrict__ mbuf,
                float* __restrict__ pbuf)
{
  const int ch = blockIdx.x;                 // 0..BATCH*HEADS*HEADS-1
  const float* e = E + (size_t)ch * SEQ;
  __shared__ float red[256];
  const int t = threadIdx.x;

  float m = -1e30f;
  for (int i = t; i < SEQ; i += 256) m = fmaxf(m, e[i] * 0.125f);
  red[t] = m; __syncthreads();
  for (int o = 128; o > 0; o >>= 1) {
    if (t < o) red[t] = fmaxf(red[t], red[t + o]);
    __syncthreads();
  }
  m = red[0]; __syncthreads();

  float p = 0.f;
  for (int i = t; i < SEQ; i += 256) p += __expf(e[i] * 0.125f - m);
  red[t] = p; __syncthreads();
  for (int o = 128; o > 0; o >>= 1) {
    if (t < o) red[t] += red[t + o];
    __syncthreads();
  }
  if (t == 0) { mbuf[ch] = m; pbuf[ch] = red[0]; }
}

// A[n,qh,s] = sum_kh exp(E/8 - m)/p
__global__ void k_attn_A(const float* __restrict__ E, const float* __restrict__ mbuf,
                         const float* __restrict__ pbuf, float* __restrict__ Ab)
{
  const size_t idx = (size_t)blockIdx.x * blockDim.x + threadIdx.x; // BATCH*HEADS*SEQ
  const int s = (int)(idx & (SEQ - 1));
  const int nq = (int)(idx >> 12);
  float a = 0.f;
  #pragma unroll
  for (int kh = 0; kh < HEADS; ++kh) {
    const int ch = nq * HEADS + kh;
    a += __expf(E[(size_t)ch * SEQ + s] * 0.125f - mbuf[ch]) / pbuf[ch];
  }
  Ab[idx] = a;
}

// attn_in[n, s', e'] with s' = qh*256 + s>>4, e' = (s&15)*64 + d  (raw reshape)
__global__ void k_build_attn_in(const float* __restrict__ Ab,
                                const float* __restrict__ Vs,
                                _Float16* __restrict__ attn_in)
{
  const size_t idx = (size_t)blockIdx.x * blockDim.x + threadIdx.x; // ROWS*EMBED
  const int ec = (int)(idx & (EMBED - 1));
  const int r  = (int)(idx >> 10);           // n*SEQ + s'
  const int n  = r >> 12, sp = r & (SEQ - 1);
  const int qh = sp >> 8;
  const int s  = ((sp & 255) << 4) | (ec >> 6);
  const int d  = ec & (HDIM - 1);
  const float val = Ab[((size_t)(n * HEADS + qh)) * SEQ + s] *
                    Vs[((size_t)(n * SEQ + s)) * HDIM + d];
  attn_in[idx] = (_Float16)val;
}

// ---------------------------- residual + LayerNorm -------------------------
__global__ __launch_bounds__(256)
void k_add_ln(const float* __restrict__ a, const float* __restrict__ b,
              const float* __restrict__ g, const float* __restrict__ be,
              float* __restrict__ outF, _Float16* __restrict__ outH)
{
  const int row = blockIdx.x;
  __shared__ float v[EMBED];
  __shared__ float red[256];
  const int t = threadIdx.x;
  const float* ar = a + (size_t)row * EMBED;
  const float* br = b + (size_t)row * EMBED;

  float sum = 0.f;
  for (int i = t; i < EMBED; i += 256) { float x = ar[i] + br[i]; v[i] = x; sum += x; }
  red[t] = sum; __syncthreads();
  for (int o = 128; o > 0; o >>= 1) { if (t < o) red[t] += red[t + o]; __syncthreads(); }
  const float mu = red[0] * (1.0f / EMBED);
  __syncthreads();

  float s2 = 0.f;
  for (int i = t; i < EMBED; i += 256) { float d = v[i] - mu; s2 += d * d; }
  red[t] = s2; __syncthreads();
  for (int o = 128; o > 0; o >>= 1) { if (t < o) red[t] += red[t + o]; __syncthreads(); }
  const float rs = rsqrtf(red[0] * (1.0f / EMBED) + LN_EPS);

  for (int i = t; i < EMBED; i += 256) {
    const float y = (v[i] - mu) * rs * g[i] + be[i];
    outF[(size_t)row * EMBED + i] = y;
    if (outH) outH[(size_t)row * EMBED + i] = (_Float16)y;
  }
}

// --------------------------------- launch ----------------------------------
extern "C" void kernel_launch(void* const* d_in, const int* in_sizes, int n_in,
                              void* d_out, int out_size, void* d_ws, size_t ws_size,
                              hipStream_t stream) {
  const float* x  = (const float*)d_in[0];
  const float* Wq = (const float*)d_in[1];
  const float* Wk = (const float*)d_in[2];
  const float* Wv = (const float*)d_in[3];
  const float* Wo = (const float*)d_in[4];
  const float* bo = (const float*)d_in[5];
  const float* W1 = (const float*)d_in[6];
  const float* b1 = (const float*)d_in[7];
  const float* W2 = (const float*)d_in[8];
  const float* b2 = (const float*)d_in[9];
  const float* g1 = (const float*)d_in[10];
  const float* be1= (const float*)d_in[11];
  const float* g2 = (const float*)d_in[12];
  const float* be2= (const float*)d_in[13];

  float* y_out = (float*)d_out;                              // [4,4096,1024]
  float* k_out = y_out + (size_t)ROWS * EMBED;               // [4,16,4096,64]
  float* v_out = k_out + (size_t)ROWS * EMBED;               // [4,16,4096,64]

  // workspace carve-out (~365 MiB, deterministic aliasing)
  char* ws = (char*)d_ws;
  size_t off = 0;
  auto alloc = [&](size_t bytes) -> char* {
    char* p = ws + off; off += (bytes + 255) & ~(size_t)255; return p;
  };
  _Float16* Wqt = (_Float16*)alloc((size_t)EMBED * EMBED * 2);
  _Float16* Wkt = (_Float16*)alloc((size_t)EMBED * EMBED * 2);
  _Float16* Wvt = (_Float16*)alloc((size_t)EMBED * EMBED * 2);
  _Float16* Wot = (_Float16*)alloc((size_t)EMBED * EMBED * 2);
  _Float16* W1t = (_Float16*)alloc((size_t)DFF * EMBED * 2);   // [4096n x 1024k]
  _Float16* W2t = (_Float16*)alloc((size_t)EMBED * DFF * 2);   // [1024n x 4096k]
  _Float16* x16 = (_Float16*)alloc((size_t)ROWS * EMBED * 2);  // aliased: attn_in
  _Float16* Q16 = (_Float16*)alloc((size_t)ROWS * EMBED * 2);  // aliased: h16
  float* Ebuf   = (float*)alloc((size_t)BATCH * HEADS * HEADS * SEQ * 4);
  float* mbuf   = (float*)alloc((size_t)BATCH * HEADS * HEADS * 4);
  float* pbuf   = (float*)alloc((size_t)BATCH * HEADS * HEADS * 4);
  float* Abuf   = (float*)alloc((size_t)BATCH * HEADS * SEQ * 4);
  float* Vsbuf  = (float*)alloc((size_t)ROWS * HDIM * 4);
  float* tmp32  = (float*)alloc((size_t)ROWS * EMBED * 4);     // attn_out, later ff2
  float* h32    = (float*)alloc((size_t)ROWS * EMBED * 4);
  _Float16* ff1 = (_Float16*)alloc((size_t)ROWS * DFF * 2);
  _Float16* attn_in = x16;   // x16 dead after V projection
  _Float16* h16     = Q16;   // Q16 dead after energy kernel
  float* ff2        = tmp32; // attn_out dead after LN1

  auto cdiv = [](long a, long b) { return (int)((a + b - 1) / b); };

  // 1) weight transpose + f16 convert
  k_transpose_f16<<<cdiv((long)EMBED*EMBED,256),256,0,stream>>>(Wq, Wqt, EMBED, EMBED);
  k_transpose_f16<<<cdiv((long)EMBED*EMBED,256),256,0,stream>>>(Wk, Wkt, EMBED, EMBED);
  k_transpose_f16<<<cdiv((long)EMBED*EMBED,256),256,0,stream>>>(Wv, Wvt, EMBED, EMBED);
  k_transpose_f16<<<cdiv((long)EMBED*EMBED,256),256,0,stream>>>(Wo, Wot, EMBED, EMBED);
  k_transpose_f16<<<cdiv((long)DFF*EMBED,256),256,0,stream>>>(W1, W1t, EMBED, DFF);
  k_transpose_f16<<<cdiv((long)DFF*EMBED,256),256,0,stream>>>(W2, W2t, DFF, EMBED);
  k_f32_to_f16<<<cdiv((long)ROWS*EMBED,256),256,0,stream>>>(x, x16, (long)ROWS*EMBED);

  const dim3 gE(EMBED/128, ROWS/128);   // N=1024 GEMMs
  const dim3 gF(DFF/128,   ROWS/128);   // N=4096 GEMM (W1)

  // 2) projections: Q -> f16 scratch; K,V -> permuted f32 directly into d_out
  k_gemm_wmma<<<gE,256,0,stream>>>(x16, Wqt, ROWS, EMBED, EMBED, nullptr, MODE_F16,  nullptr, Q16);
  k_gemm_wmma<<<gE,256,0,stream>>>(x16, Wkt, ROWS, EMBED, EMBED, nullptr, MODE_PERM, k_out,   nullptr);
  k_gemm_wmma<<<gE,256,0,stream>>>(x16, Wvt, ROWS, EMBED, EMBED, nullptr, MODE_PERM, v_out,   nullptr);

  // 3) degenerate attention
  k_energy<<<ROWS,256,0,stream>>>(Q16, k_out, v_out, Ebuf, Vsbuf);
  k_sm_stats<<<BATCH*HEADS*HEADS,256,0,stream>>>(Ebuf, mbuf, pbuf);
  k_attn_A<<<cdiv((long)BATCH*HEADS*SEQ,256),256,0,stream>>>(Ebuf, mbuf, pbuf, Abuf);
  k_build_attn_in<<<cdiv((long)ROWS*EMBED,256),256,0,stream>>>(Abuf, Vsbuf, attn_in);

  // 4) output projection + residual LN1
  k_gemm_wmma<<<gE,256,0,stream>>>(attn_in, Wot, ROWS, EMBED, EMBED, bo, MODE_F32, tmp32, nullptr);
  k_add_ln<<<ROWS,256,0,stream>>>(tmp32, x, g1, be1, h32, h16);

  // 5) FFN + residual LN2 -> y
  k_gemm_wmma<<<gF,256,0,stream>>>(h16, W1t, ROWS, DFF, EMBED, b1, MODE_RELU16, nullptr, ff1);
  k_gemm_wmma<<<gE,256,0,stream>>>(ff1, W2t, ROWS, EMBED, DFF, b2, MODE_F32, ff2, nullptr);
  k_add_ln<<<ROWS,256,0,stream>>>(ff2, h32, g2, be2, y_out, nullptr);

  (void)in_sizes; (void)n_in; (void)out_size; (void)ws_size;
}
